// BaseModel_86930138071100
// MI455X (gfx1250) — matrix operand order repaired and checked by reading, not compiled, not run on hardware
//
#include <hip/hip_runtime.h>

// ColBERT late-interaction MaxSim on gfx1250 (MI455X), compute-bound -> WMMA.
// f32 inputs are split into bf16 hi+lo in LDS; each dot is computed as
// AhiBhi + AhiBlo + AloBhi on v_wmma_f32_16x16x32_bf16 (f32 accumulate),
// recovering ~f32 accuracy on the high-throughput bf16 matrix pipe.

typedef __attribute__((ext_vector_type(16))) __bf16 v16bf;
typedef __attribute__((ext_vector_type(8)))  float  v8f;
typedef __attribute__((ext_vector_type(4)))  unsigned int u32x4;

namespace {
constexpr int NQ   = 32;   // queries
constexpr int MTOK = 32;   // query tokens
constexpr int ND   = 256;  // docs
constexpr int NTOK = 180;  // doc tokens
constexpr int NPAD = 192;  // doc tokens padded to 12 WMMA tiles
constexpr int H    = 128;  // head dim
constexpr int ROW  = 136;  // LDS row stride in bf16 elems (272B, bank-conflict-free)
}

union V16U { v16bf v; u32x4 u[2]; };

__device__ __forceinline__ unsigned short f2bf(float f) {
  unsigned int u = __float_as_uint(f);
  u += 0x7FFFu + ((u >> 16) & 1u);          // round-to-nearest-even
  return (unsigned short)(u >> 16);
}
__device__ __forceinline__ float bf2f(unsigned short h) {
  return __uint_as_float(((unsigned int)h) << 16);
}

// A operand (16x32 bf16): lane holds 16 contiguous K values (matches B)
__device__ __forceinline__ v16bf load_a(const unsigned short* p) {
  V16U r;
  r.u[0] = *(const u32x4*)(p);
  r.u[1] = *(const u32x4*)(p + 8);
  return r.v;
}
// B operand (32x16 bf16): lane holds 16 contiguous K values
__device__ __forceinline__ v16bf load_b(const unsigned short* p) {
  V16U r;
  r.u[0] = *(const u32x4*)(p);
  r.u[1] = *(const u32x4*)(p + 8);
  return r.v;
}
__device__ __forceinline__ v8f wmma_bf16(v16bf a, v16bf b, v8f c) {
  return __builtin_amdgcn_wmma_f32_16x16x32_bf16(false, a, false, b, (short)0, c,
                                                 false, false);
}

__global__ __launch_bounds__(256)
void colbert_maxsim_wmma(const float* __restrict__ Qg, const float* __restrict__ Dg,
                         const float* __restrict__ qmask, const float* __restrict__ dmask,
                         float* __restrict__ out) {
  extern __shared__ __align__(16) unsigned char smem[];
  unsigned short* sDhi = (unsigned short*)smem;            // NPAD*ROW
  unsigned short* sDlo = sDhi + NPAD * ROW;
  unsigned short* sQhi = sDlo + NPAD * ROW;                // MTOK*ROW
  unsigned short* sQlo = sQhi + MTOK * ROW;
  float*          wmax = (float*)(sQlo + MTOK * ROW);      // 8 waves * 16 rows

  const int tid   = threadIdx.x;
  const int d     = blockIdx.x;
  const int qbase = blockIdx.y * 16;

  // ---- stage D[d] (masked, zero-padded to 192 rows) as split bf16 in LDS ----
  for (int i = tid; i < NPAD * H; i += 256) {
    const int n = i >> 7, k = i & (H - 1);
    float x = 0.0f;
    if (n < NTOK) x = Dg[(d * NTOK + n) * H + k] * dmask[d * NTOK + n];
    const unsigned short hi = f2bf(x);
    const unsigned short lo = f2bf(x - bf2f(hi));
    sDhi[n * ROW + k] = hi;
    sDlo[n * ROW + k] = lo;
  }

  const int lane = tid & 31;
  const int w    = tid >> 5;
  const int half = lane >> 4;     // K-phase select for A/B operand layout
  const int col  = lane & 15;
  const int mt   = w >> 2;        // which 16 query-token rows (0..1)
  const int nt0  = 3 * (w & 3);   // first of this wave's 3 doc-token tiles

  const int aoff = (mt * 16 + col) * ROW + half * 16;  // + ks*32
  const int boff = (nt0 * 16 + col) * ROW + half * 16; // + ks*32 (+16*ROW, +32*ROW)

  for (int qi = 0; qi < 16; ++qi) {
    const int q = qbase + qi;
    __syncthreads();

    // ---- stage Q[q] (masked) as split bf16 ----
    for (int i = tid; i < MTOK * H; i += 256) {
      const int m = i >> 7, k = i & (H - 1);
      const float x = Qg[(q * MTOK + m) * H + k] * qmask[q * MTOK + m];
      const unsigned short hi = f2bf(x);
      const unsigned short lo = f2bf(x - bf2f(hi));
      sQhi[m * ROW + k] = hi;
      sQlo[m * ROW + k] = lo;
    }
    __syncthreads();

    // ---- 32x192 sim matrix: 3 C tiles per wave, K=128 in 4 steps of 32 ----
    v8f c0 = {}; v8f c1 = {}; v8f c2 = {};
#pragma unroll
    for (int ks = 0; ks < 4; ++ks) {
      const int kb = ks * 32;
      const v16bf ahi = load_a(sQhi + aoff + kb);
      const v16bf alo = load_a(sQlo + aoff + kb);

      v16bf bhi = load_b(sDhi + boff + kb);
      v16bf blo = load_b(sDlo + boff + kb);
      c0 = wmma_bf16(ahi, bhi, c0);
      c0 = wmma_bf16(ahi, blo, c0);
      c0 = wmma_bf16(alo, bhi, c0);

      bhi = load_b(sDhi + boff + 16 * ROW + kb);
      blo = load_b(sDlo + boff + 16 * ROW + kb);
      c1 = wmma_bf16(ahi, bhi, c1);
      c1 = wmma_bf16(ahi, blo, c1);
      c1 = wmma_bf16(alo, bhi, c1);

      bhi = load_b(sDhi + boff + 32 * ROW + kb);
      blo = load_b(sDlo + boff + 32 * ROW + kb);
      c2 = wmma_bf16(ahi, bhi, c2);
      c2 = wmma_bf16(ahi, blo, c2);
      c2 = wmma_bf16(alo, bhi, c2);
    }

    // ---- per-row max over this wave's 48 doc tokens (exclude padded n>=180) ----
    const int n0 = nt0 * 16 + col;
#pragma unroll
    for (int r = 0; r < 8; ++r) {
      float v = -3.0e38f;
      if (n0 < NTOK)      v = fmaxf(v, c0[r]);
      if (n0 + 16 < NTOK) v = fmaxf(v, c1[r]);
      if (n0 + 32 < NTOK) v = fmaxf(v, c2[r]);
      v = fmaxf(v, __shfl_xor(v, 1, 32));   // reduce across the 16-lane half
      v = fmaxf(v, __shfl_xor(v, 2, 32));
      v = fmaxf(v, __shfl_xor(v, 4, 32));
      v = fmaxf(v, __shfl_xor(v, 8, 32));
      if (col == 0) wmax[w * 16 + half * 8 + r] = v;  // row m = mt*16 + 8*half + r
    }
    __syncthreads();

    // ---- combine 4 waves per row group, sum 32 rows, emit score ----
    if (tid < 32) {
      const int grp = tid >> 4, row = tid & 15;
      float rm = fmaxf(fmaxf(wmax[(grp * 4 + 0) * 16 + row],
                             wmax[(grp * 4 + 1) * 16 + row]),
                       fmaxf(wmax[(grp * 4 + 2) * 16 + row],
                             wmax[(grp * 4 + 3) * 16 + row]));
      rm += __shfl_xor(rm, 1, 32);
      rm += __shfl_xor(rm, 2, 32);
      rm += __shfl_xor(rm, 4, 32);
      rm += __shfl_xor(rm, 8, 32);
      rm += __shfl_xor(rm, 16, 32);
      if (tid == 0) out[q * ND + d] = rm;
    }
  }
}

extern "C" void kernel_launch(void* const* d_in, const int* in_sizes, int n_in,
                              void* d_out, int out_size, void* d_ws, size_t ws_size,
                              hipStream_t stream) {
  (void)in_sizes; (void)n_in; (void)out_size; (void)d_ws; (void)ws_size;
  const float* Q  = (const float*)d_in[0];
  const float* D  = (const float*)d_in[1];
  const float* qm = (const float*)d_in[2];
  const float* dm = (const float*)d_in[3];
  float* out = (float*)d_out;

  const size_t shmem =
      (size_t)(2 * NPAD * ROW + 2 * MTOK * ROW) * sizeof(unsigned short) +
      (size_t)(8 * 16) * sizeof(float);   // ~120 KB: CDNA5 WGP has 320 KB LDS

  // Opt in to >64KB dynamic LDS (idempotent; harmless if default allows it).
  (void)hipFuncSetAttribute(reinterpret_cast<const void*>(&colbert_maxsim_wmma),
                            hipFuncAttributeMaxDynamicSharedMemorySize,
                            (int)shmem);

  dim3 grid(ND, NQ / 16);
  colbert_maxsim_wmma<<<grid, dim3(256), shmem, stream>>>(Q, D, qm, dm, out);
}
